// ItemVectorTransform_81458349736120
// MI455X (gfx1250) — matrix operand order repaired and checked by reading, not compiled, not run on hardware
//
#include <hip/hip_runtime.h>

typedef __attribute__((ext_vector_type(16))) _Float16 v16h;
typedef __attribute__((ext_vector_type(2)))  _Float16 v2h;
typedef __attribute__((ext_vector_type(16))) __bf16   v16bf;
typedef __attribute__((ext_vector_type(2)))  __bf16   v2bf;
typedef __attribute__((ext_vector_type(8)))  float    v8f;

#define B_ROWS   2048
#define K_SLOTS  100000
#define D_DIM    50
#define NW       8                       // waves per block (wave32)
#define TILE_M   16                      // rows of x per workgroup
#define CHUNK    32                      // memory slots per K-chunk
#define NCHUNK   (K_SLOTS / CHUNK)       // 3125, exact
#define NG16     (K_SLOTS / 16)          // 6250 slot-16 groups (S-pack)
#define N_RT     (B_ROWS / TILE_M)       // 128 row tiles
#define LOG2E    1.4426950408889634f     // folded into x so exp(s) == exp2(s')

// ---- packed fragment workspace layout (units: 2-byte elements) ----
// S-pack : f16  B^T fragments for score GEMM  [group g][dfrag f][lane][16]
// O-pack : bf16 B   fragments for readout GEMM[chunk h][dtile t][lane][16]
// X-pack : f16  A   fragments of x * log2(e)  [rowtile][kfrag f][lane][16]
#define HALVES_S ((size_t)NG16   * 2 * 32 * 16)   // 6,400,000
#define HALVES_O ((size_t)NCHUNK * 4 * 32 * 16)   // 6,400,000
#define HALVES_X ((size_t)N_RT   * 2 * 32 * 16)   //   131,072
#define OFF_S    ((size_t)0)
#define OFF_O    (HALVES_S)
#define OFF_X    (HALVES_S + HALVES_O)
#define WS_BYTES ((HALVES_S + HALVES_O + HALVES_X) * 2)  // ~25.9 MB

__device__ __forceinline__ float fast_exp2(float v) {
#if __has_builtin(__builtin_amdgcn_exp2f)
  return __builtin_amdgcn_exp2f(v);      // bare v_exp_f32
#else
  return exp2f(v);
#endif
}

__device__ __forceinline__ float half_red_sum16(float v) {
  v += __shfl_xor(v, 1, 32);
  v += __shfl_xor(v, 2, 32);
  v += __shfl_xor(v, 4, 32);
  v += __shfl_xor(v, 8, 32);
  return v;
}

// ======================= pack kernels (run once per launch) =======================
// Each thread emits one 32-bit word = 2 consecutive 2-byte elements of a fragment.
// word index wi: j2 = wi&7 (element pair), lane = (wi>>3)&31, frag idx = wi>>8.

__global__ void pack_S_kernel(const float* __restrict__ M, _Float16* __restrict__ dst) {
  size_t wi = (size_t)blockIdx.x * 256 + threadIdx.x;
  if (wi >= HALVES_S / 2) return;
  int j2 = (int)(wi & 7), lane = (int)((wi >> 3) & 31);
  size_t idx = wi >> 8;
  int f = (int)(idx & 1); size_t g = idx >> 1;
  int col = lane & 15, hw = lane >> 4;
  size_t row = 16 * g + col;                 // slot (always < K)
  int d0 = 32 * f + 16 * hw + 2 * j2;        // B^T row = feature dim
  float v0 = (d0     < D_DIM) ? M[row * D_DIM + d0]     : 0.f;
  float v1 = (d0 + 1 < D_DIM) ? M[row * D_DIM + d0 + 1] : 0.f;
  ((v2h*)dst)[wi] = (v2h){(_Float16)v0, (_Float16)v1};
}

__global__ void pack_O_kernel(const float* __restrict__ M, __bf16* __restrict__ dst) {
  size_t wi = (size_t)blockIdx.x * 256 + threadIdx.x;
  if (wi >= HALVES_O / 2) return;
  int j2 = (int)(wi & 7), lane = (int)((wi >> 3) & 31);
  size_t idx = wi >> 8;
  int t = (int)(idx & 3); size_t h = idx >> 2;
  int col = lane & 15, hw = lane >> 4;
  int dcol = 16 * t + col;                   // B col = feature dim
  int dc = dcol < D_DIM ? dcol : D_DIM - 1;
  float fm = (dcol < D_DIM) ? 1.f : 0.f;
  size_t r0 = 32 * h + 16 * hw + 2 * j2;     // B row = slot (always < K)
  float v0 = M[r0 * D_DIM + dc] * fm;
  float v1 = M[(r0 + 1) * D_DIM + dc] * fm;
  ((v2bf*)dst)[wi] = (v2bf){(__bf16)v0, (__bf16)v1};
}

__global__ void pack_X_kernel(const float* __restrict__ x, _Float16* __restrict__ dst) {
  size_t wi = (size_t)blockIdx.x * 256 + threadIdx.x;
  if (wi >= HALVES_X / 2) return;
  int j2 = (int)(wi & 7), lane = (int)((wi >> 3) & 31);
  size_t idx = wi >> 8;
  int f = (int)(idx & 1); size_t rt = idx >> 1;
  int col = lane & 15, hw = lane >> 4;
  int j0 = 2 * j2;
  int k0 = 32 * f + 16 * hw + j0;  // A K index (contiguous order)
  size_t row = 16 * rt + col;
  float v0 = (k0     < D_DIM) ? x[row * D_DIM + k0] * LOG2E : 0.f;
  float v1 = (k0 + 1 < D_DIM) ? x[row * D_DIM + k0 + 1] * LOG2E : 0.f;
  ((v2h*)dst)[wi] = (v2h){(_Float16)v0, (_Float16)v1};
}

// ======================= fused attention readout =======================
// Max-free softmax: scores are dots of 50 unit-normal pairs (|s| <~ 50), so
// exp(s) stays inside f32/bf16 dynamic range. log2(e) is folded into the x
// fragments so exp(s) = exp2(s') lowers to a bare v_exp_f32. P flows to the
// readout GEMM in bf16 (f32-range exponent); score GEMM operands stay f16
// (better mantissa before exp). No per-chunk cross-lane reductions, no
// accumulator rescaling.
template <bool PACKED>
__global__ __launch_bounds__(NW * 32)
void attn_main(const float* __restrict__ x, const float* __restrict__ M,
               const _Float16* __restrict__ pS, const __bf16* __restrict__ pO,
               const _Float16* __restrict__ pX, float* __restrict__ out)
{
  __shared__ __bf16 ldsP[NW][TILE_M][CHUNK];         // 8 KB (P transpose staging)
  __shared__ float  red_l[NW][TILE_M];               // cross-wave partial sums
  __shared__ float  red_O[NW][TILE_M][64];           // 32 KB

  const int tid  = threadIdx.x;
  const int w    = tid >> 5;
  const int lane = tid & 31;
  const int hw   = (lane >> 4) & 1;
  const int col  = lane & 15;
  const int rt   = blockIdx.x;
  const int b0   = rt * TILE_M;

  // ---- A fragments of x*log2e (16 rows x padded-64, two 16x32 f16 frags) ----
  v16h ax[2];
  if (PACKED) {
#pragma unroll
    for (int f = 0; f < 2; ++f)
      ax[f] = *(const v16h*)(pX + ((size_t)(rt * 2 + f) * 32 + lane) * 16);
  } else {
    const float* xr = x + (size_t)(b0 + col) * D_DIM;
#pragma unroll
    for (int f = 0; f < 2; ++f)
#pragma unroll
      for (int j = 0; j < 16; ++j) {
        int k = 32 * f + 16 * hw + j;
        ax[f][j] = (_Float16)(xr[k < D_DIM ? k : D_DIM - 1] *
                              (k < D_DIM ? LOG2E : 0.f));
      }
  }

  float lrow[8];                        // per-lane partial softmax denominators
  v8f o[4] = {};                        // O accumulator: 16 x 64 (4 col tiles)
#pragma unroll
  for (int r = 0; r < 8; ++r) lrow[r] = 0.f;

  // ============ streaming K loop: 32 slots / chunk, waves split K ============
  for (int c = w; c < NCHUNK; c += NW) {
    const int kk0 = c * CHUNK;

    {   // warm caches for this wave's next chunk -> global_prefetch_b8
      int nc = (c + NW < NCHUNK) ? c + NW : c;
      if (PACKED) {
        __builtin_prefetch(pS + (size_t)(2 * nc) * 2 * 32 * 16, 0, 1);
        __builtin_prefetch(pO + (size_t)nc * 4 * 32 * 16, 0, 1);
      } else {
        __builtin_prefetch(M + (size_t)(nc * CHUNK) * D_DIM, 0, 1);
      }
    }

    // ---- S tiles: S[:,16nt..] = (x*log2e) @ Mc^T (f16 WMMA, f32 accum) ----
    v8f s[2];
#pragma unroll
    for (int nt = 0; nt < 2; ++nt) {
      v16h bs0, bs1;
      if (PACKED) {
        const size_t g = (size_t)(2 * c + nt) * 2;
        bs0 = *(const v16h*)(pS + ((g + 0) * 32 + lane) * 16);
        bs1 = *(const v16h*)(pS + ((g + 1) * 32 + lane) * 16);
      } else {
        const float* mr = M + (size_t)(kk0 + 16 * nt + col) * D_DIM;
#pragma unroll
        for (int j = 0; j < 16; ++j) {
          int d0 = 16 * hw + j, d1 = 32 + 16 * hw + j;
          bs0[j] = (_Float16)(mr[d0]);   // d0 <= 31 < 50, always valid
          bs1[j] = (_Float16)(mr[d1 < D_DIM ? d1 : D_DIM - 1] * (d1 < D_DIM ? 1.f : 0.f));
        }
      }
      v8f acc = {};
      acc = __builtin_amdgcn_wmma_f32_16x16x32_f16(false, ax[0], false, bs0,
                                                   (short)0, acc, false, false);
      acc = __builtin_amdgcn_wmma_f32_16x16x32_f16(false, ax[1], false, bs1,
                                                   (short)0, acc, false, false);
      s[nt] = acc;
    }

    // ---- P = exp2(S'); accumulate denominator per-lane; stage P in bf16 ----
#pragma unroll
    for (int r = 0; r < 8; ++r) {
      float p0 = fast_exp2(s[0][r]);
      float p1 = fast_exp2(s[1][r]);
      lrow[r] += p0 + p1;
      ldsP[w][r + 8 * hw][col]      = (__bf16)p0;   // C-layout -> LDS
      ldsP[w][r + 8 * hw][16 + col] = (__bf16)p1;
    }

    // ---- read P back in A-layout (16x32, all K used) ----
    asm volatile("s_wait_dscnt 0x0" ::: "memory");    // wave-private RAW fence
    v16bf pa;
#pragma unroll
    for (int j = 0; j < 16; ++j)
      pa[j] = ldsP[w][col][16 * hw + j];

    // ---- O[16x64] += P @ Mc : 4 col tiles, one full K=32 bf16 WMMA each ----
#pragma unroll
    for (int t = 0; t < 4; ++t) {
      v16bf bo;
      if (PACKED) {
        bo = *(const v16bf*)(pO + (((size_t)c * 4 + t) * 32 + lane) * 16);
      } else {
        const int dcol = 16 * t + col;
        const int dc   = dcol < D_DIM ? dcol : D_DIM - 1;
        const float fm = (dcol < D_DIM) ? 1.f : 0.f;
#pragma unroll
        for (int j = 0; j < 16; ++j)
          bo[j] = (__bf16)(M[(size_t)(kk0 + 16 * hw + j) * D_DIM + dc] * fm);
      }
      o[t] = __builtin_amdgcn_wmma_f32_16x16x32_bf16(false, pa, false, bo,
                                                     (short)0, o[t], false, false);
    }
  }

  // ================= cross-wave merge: plain sums =================
#pragma unroll
  for (int r = 0; r < 8; ++r) {
    float lsum = half_red_sum16(lrow[r]);   // one-time cross-lane reduce
    if (col == 0) red_l[w][r + 8 * hw] = lsum;
#pragma unroll
    for (int t = 0; t < 4; ++t)
      red_O[w][r + 8 * hw][16 * t + col] = o[t][r];
  }
  __syncthreads();

  for (int e = tid; e < TILE_M * 64; e += NW * 32) {
    int row = e >> 6, d = e & 63;
    if (d >= D_DIM) continue;
    float L = 0.f, val = 0.f;
#pragma unroll
    for (int ww = 0; ww < NW; ++ww) {
      L   += red_l[ww][row];
      val += red_O[ww][row][d];
    }
    out[(size_t)(b0 + row) * (2 * D_DIM) + D_DIM + d] = val / L;
  }

  for (int e = tid; e < TILE_M * D_DIM; e += NW * 32) {
    int row = e / D_DIM, j = e % D_DIM;
    out[(size_t)(b0 + row) * (2 * D_DIM) + j] = x[(size_t)(b0 + row) * D_DIM + j];
  }
}

extern "C" void kernel_launch(void* const* d_in, const int* in_sizes, int n_in,
                              void* d_out, int out_size, void* d_ws, size_t ws_size,
                              hipStream_t stream) {
  const float* x = (const float*)d_in[0];   // [2048, 50]
  const float* M = (const float*)d_in[1];   // [100000, 50]
  float* out = (float*)d_out;               // [2048, 100]
  (void)in_sizes; (void)n_in; (void)out_size;

  dim3 block(NW * 32);
  if (d_ws != nullptr && ws_size >= WS_BYTES) {
    _Float16* base = (_Float16*)d_ws;
    _Float16* pS = base + OFF_S;
    __bf16*   pO = (__bf16*)(base + OFF_O);
    _Float16* pX = base + OFF_X;
    pack_S_kernel<<<dim3((unsigned)((HALVES_S / 2 + 255) / 256)), 256, 0, stream>>>(M, pS);
    pack_O_kernel<<<dim3((unsigned)((HALVES_O / 2 + 255) / 256)), 256, 0, stream>>>(M, pO);
    pack_X_kernel<<<dim3((unsigned)((HALVES_X / 2 + 255) / 256)), 256, 0, stream>>>(x, pX);
    attn_main<true><<<dim3(N_RT), block, 0, stream>>>(x, M, pS, pO, pX, out);
  } else {
    attn_main<false><<<dim3(N_RT), block, 0, stream>>>(x, M, nullptr, nullptr, nullptr, out);
  }
}
